// MaxPoolSpatialAttentionModule_21320217657666
// MI455X (gfx1250) — matrix operand order repaired and checked by reading, not compiled, not run on hardware
//
#include <hip/hip_runtime.h>
#include <hip/hip_bf16.h>
#include <math.h>

typedef __bf16 bf16_t;
typedef __attribute__((ext_vector_type(16))) __bf16 v16bf;
typedef __attribute__((ext_vector_type(8)))  float  v8f;
typedef __attribute__((ext_vector_type(2)))  float  v2f;

namespace {
constexpr int BATCH = 16, C = 256, CR = 16, H = 56, W = 56;
constexpr int HW  = H * W;            // 3136
constexpr int HP = 28, WP = 28, HWP = HP * WP;  // 784
constexpr int KP  = 800;              // 784 padded to 25*32 for bf16 WMMA K-loop
constexpr float BN_EPS = 1e-5f;

// workspace layout (bytes)
constexpr size_t OFF_C2  = 0;                                   // f32 [B][CR][HW]
constexpr size_t SZ_C2   = (size_t)BATCH * CR * HW * 4;
constexpr size_t OFF_C1P = OFF_C2 + SZ_C2;                      // f32 [B][CR][HWP]
constexpr size_t SZ_C1P  = (size_t)BATCH * CR * HWP * 4;
constexpr size_t OFF_C3P = OFF_C1P + SZ_C1P;                    // bf16 [B][C][KP], cols 784..799 zero
}

__device__ __forceinline__ v8f zero8() {
    v8f v;
#pragma unroll
    for (int i = 0; i < 8; ++i) v[i] = 0.f;
    return v;
}

// ---------------------------------------------------------------------------
// Async global->LDS staging (CDNA5 GLOBAL_LOAD_ASYNC_TO_LDS_B128, ASYNCcnt),
// with a synchronous fallback producing the identical LDS layout.
// Builtin prototype (from hipcc diagnostic): (v4i as1*, v4i as3*, imm int, imm int)
// ---------------------------------------------------------------------------
#if defined(__AMDGCN__) && __has_builtin(__builtin_amdgcn_global_load_async_to_lds_b128)
#define HAVE_ASYNC_LDS 1
typedef int v4i_g __attribute__((vector_size(16)));
typedef __attribute__((address_space(1))) v4i_g* as1_v4i_ptr;
typedef __attribute__((address_space(3))) v4i_g* as3_v4i_ptr;
__device__ __forceinline__ void async_cp16(const float* g, float* l) {
    __builtin_amdgcn_global_load_async_to_lds_b128(
        (as1_v4i_ptr)(void*)(const_cast<float*>(g)),
        (as3_v4i_ptr)(void*)l, 0, 0);
}
__device__ __forceinline__ void wait_async_lds() {
    asm volatile("s_wait_asynccnt 0x0" ::: "memory");
}
#else
#define HAVE_ASYNC_LDS 0
__device__ __forceinline__ void wait_async_lds() {}
#endif

// Stage a [32 k][112 col] f32 tile. gsrc already offset to (k0, n0); row stride HW floats.
__device__ __forceinline__ void stage_tile(const float* __restrict__ gsrc,
                                           float* ldst, int tid) {
#if HAVE_ASYNC_LDS
#pragma unroll
    for (int i = 0; i < 7; ++i) {               // 896 16B chunks / 128 threads
        const int c = tid + 128 * i;
        const int row = c / 28, off = (c % 28) * 4;   // 28 chunks of 4 floats per row
        async_cp16(gsrc + (size_t)row * HW + off, ldst + row * 112 + off);
    }
#else
    for (int idx = tid; idx < 32 * 112; idx += 128) {
        const int kk = idx / 112, col = idx % 112;
        ldst[kk * 112 + col] = gsrc[(size_t)kk * HW + col];
    }
#endif
}

// A fragment (16x32 bf16, M x K) built from an f32 weight row (one row per lane, M = lane%16).
// K pattern: lanes<16 -> K = {0..15}; lanes>=16 -> K = {16..31} (relative to k0).
__device__ __forceinline__ v16bf load_a_w(const float* wrow, int k0, int lane) {
    const int h = lane >> 4;
    const float* p0 = wrow + k0 + 16 * h;
    const float* p1 = p0 + 8;
    v16bf a;
#pragma unroll
    for (int e = 0; e < 8; ++e) a[e] = (bf16_t)p0[e];
#pragma unroll
    for (int e = 0; e < 8; ++e) a[8 + e] = (bf16_t)p1[e];
    return a;
}

// B fragment (32x16 bf16, K x N) from an f32 LDS tile [32 k][112 col], converting on the fly.
// lane: N = lane%16 (-> col), K = e + 16*(lane>=16).
__device__ __forceinline__ v16bf load_b_xf(const float* xf, int col, int lane) {
    const int hh = (lane >> 4) * 16;
    v16bf b;
#pragma unroll
    for (int e = 0; e < 16; ++e)
        b[e] = (bf16_t)xf[(e + hh) * 112 + col];
    return b;
}

// A fragment (16x32 bf16) from a global bf16 row with stride KP (row already selected per lane).
__device__ __forceinline__ v16bf load_a_row(const bf16_t* row, int k0, int lane) {
    union { uint4 q[2]; v16bf v; } u;
    const bf16_t* p = row + k0 + 8 * (lane >> 4);
    u.q[0] = *(const uint4*)p;
    u.q[1] = *(const uint4*)(p + 16);
    return u.v;
}

// B fragment (32x16 bf16) = attn^T tile from LDS attn[16 rows][KP cols].
__device__ __forceinline__ v16bf load_b_attn(const bf16_t* attn, int k0, int lane) {
    union { uint4 q[2]; v16bf v; } u;
    const bf16_t* p = attn + (size_t)(lane & 15) * KP + k0 + 8 * (lane >> 4);
    u.q[0] = *(const uint4*)p;
    u.q[1] = *(const uint4*)(p + 16);
    return u.v;
}

// ---------------------------------------------------------------------------
// Kernel 1: branches 1+2. GEMM [32 x 256] x [256 x 112] per (batch, chunk).
// Rows 0..15 = w1 -> BN1+PReLU -> 2x2 maxpool -> c1p ; rows 16..31 = w2 -> BN2+PReLU -> c2.
// Double-buffered async staging of the x tile.
// ---------------------------------------------------------------------------
__global__ __launch_bounds__(128) void conv12_kernel(
    const float* __restrict__ x,
    const float* __restrict__ w1, const float* __restrict__ w2,
    const float* __restrict__ bn1w, const float* __restrict__ bn1b,
    const float* __restrict__ bn1m, const float* __restrict__ bn1v,
    const float* __restrict__ p1,
    const float* __restrict__ bn2w, const float* __restrict__ bn2b,
    const float* __restrict__ bn2m, const float* __restrict__ bn2v,
    const float* __restrict__ p2,
    float* __restrict__ c2out, float* __restrict__ c1p) {
    __shared__ alignas(16) float xf[2][32 * 112];   // double-buffered x tile (f32)
    __shared__ float ldsc1[16 * 112];               // branch-1 activations for pooling

    const int b = blockIdx.y, chunk = blockIdx.x;
    const int n0 = chunk * 112;                     // columns for image rows {2*chunk, 2*chunk+1}
    const int tid = threadIdx.x, lane = tid & 31, wave = tid >> 5;
    const int nl = lane & 15, hh = lane >> 4;
    const float* xB = x + (size_t)b * C * HW + n0;

    v8f acc[4];
#pragma unroll
    for (int s = 0; s < 4; ++s) acc[s] = zero8();

    stage_tile(xB, xf[0], tid);                     // prologue: k0 = 0
    int cur = 0;
    for (int k0 = 0; k0 < C; k0 += 32) {
        wait_async_lds();                           // this wave's copies for xf[cur] done
        __syncthreads();                            // all waves' copies done; prev reads done
        if (k0 + 32 < C) stage_tile(xB + (size_t)(k0 + 32) * HW, xf[cur ^ 1], tid);
#pragma unroll
        for (int s = 0; s < 4; ++s) {
            const int t = wave + 4 * s;             // 14 tiles: mt = t/7, nt = t%7
            if (t < 14) {
                const int mt = t / 7, nt = t % 7;
                const float* wrow = (mt == 0 ? w1 : w2) + (size_t)nl * C;
                v16bf a = load_a_w(wrow, k0, lane);
                v16bf bb = load_b_xf(xf[cur], nt * 16 + nl, lane);
                acc[s] = __builtin_amdgcn_wmma_f32_16x16x32_bf16(
                    false, a, false, bb, (short)0, acc[s], false, false);
            }
        }
        cur ^= 1;
    }

#pragma unroll
    for (int s = 0; s < 4; ++s) {
        const int t = wave + 4 * s;
        if (t >= 14) continue;
        const int mt = t / 7, nt = t % 7;
#pragma unroll
        for (int r = 0; r < 8; ++r) {
            const int m = r + 8 * hh;               // channel within branch (0..15)
            float sc, sh, pr;
            if (mt == 0) {
                sc = bn1w[m] / sqrtf(bn1v[m] + BN_EPS);
                sh = bn1b[m] - bn1m[m] * sc;
                pr = p1[m];
            } else {
                sc = bn2w[m] / sqrtf(bn2v[m] + BN_EPS);
                sh = bn2b[m] - bn2m[m] * sc;
                pr = p2[m];
            }
            float v = acc[s][r] * sc + sh;
            v = v >= 0.f ? v : pr * v;
            if (mt == 0)
                ldsc1[m * 112 + nt * 16 + nl] = v;
            else
                c2out[(size_t)b * CR * HW + (size_t)m * HW + n0 + nt * 16 + nl] = v;
        }
    }
    __syncthreads();
    // 2x2 maxpool of branch 1: chunk covers image rows 2*chunk (cols 0..55) and 2*chunk+1 (cols 56..111)
    for (int i = tid; i < 16 * 28; i += 128) {
        const int m = i / 28, wp = i % 28;
        const float* rowp = ldsc1 + m * 112;
        float v = fmaxf(fmaxf(rowp[2 * wp], rowp[2 * wp + 1]),
                        fmaxf(rowp[56 + 2 * wp], rowp[56 + 2 * wp + 1]));
        c1p[(size_t)b * CR * HWP + (size_t)m * HWP + chunk * 28 + wp] = v;
    }
}

// ---------------------------------------------------------------------------
// Kernel 2: branch 3. GEMM [16 x 256] x [256 x 112] per (batch, mtile, chunk),
// BN3+PReLU, fused 2x2 maxpool -> c3p (bf16, K padded to KP with zeros).
// ---------------------------------------------------------------------------
__global__ __launch_bounds__(128) void conv3_kernel(
    const float* __restrict__ x, const float* __restrict__ w3,
    const float* __restrict__ bn3w, const float* __restrict__ bn3b,
    const float* __restrict__ bn3m, const float* __restrict__ bn3v,
    const float* __restrict__ p3,
    bf16_t* __restrict__ c3p) {
    __shared__ alignas(16) float xf[2][32 * 112];
    __shared__ float ldsc[16 * 112];

    const int b = blockIdx.z, mtile = blockIdx.y, chunk = blockIdx.x;
    const int n0 = chunk * 112;
    const int tid = threadIdx.x, lane = tid & 31, wave = tid >> 5;
    const int nl = lane & 15, hh = lane >> 4;
    const float* xB = x + (size_t)b * C * HW + n0;

    v8f acc[2];
    acc[0] = zero8();
    acc[1] = zero8();

    stage_tile(xB, xf[0], tid);
    int cur = 0;
    for (int k0 = 0; k0 < C; k0 += 32) {
        wait_async_lds();
        __syncthreads();
        if (k0 + 32 < C) stage_tile(xB + (size_t)(k0 + 32) * HW, xf[cur ^ 1], tid);
        const float* wrow = w3 + (size_t)(mtile * 16 + nl) * C;
        v16bf a = load_a_w(wrow, k0, lane);
#pragma unroll
        for (int s = 0; s < 2; ++s) {
            const int nt = wave + 4 * s;            // 7 n-tiles over 4 waves
            if (nt < 7) {
                v16bf bb = load_b_xf(xf[cur], nt * 16 + nl, lane);
                acc[s] = __builtin_amdgcn_wmma_f32_16x16x32_bf16(
                    false, a, false, bb, (short)0, acc[s], false, false);
            }
        }
        cur ^= 1;
    }

#pragma unroll
    for (int s = 0; s < 2; ++s) {
        const int nt = wave + 4 * s;
        if (nt >= 7) continue;
#pragma unroll
        for (int r = 0; r < 8; ++r) {
            const int m = r + 8 * hh;
            const int ch = mtile * 16 + m;
            const float sc = bn3w[ch] / sqrtf(bn3v[ch] + BN_EPS);
            const float sh = bn3b[ch] - bn3m[ch] * sc;
            float v = acc[s][r] * sc + sh;
            v = v >= 0.f ? v : p3[ch] * v;
            ldsc[m * 112 + nt * 16 + nl] = v;
        }
    }
    __syncthreads();
    bf16_t* c3pB = c3p + (size_t)b * C * KP;
    for (int i = tid; i < 16 * 28; i += 128) {
        const int m = i / 28, wp = i % 28;
        const float* rowp = ldsc + m * 112;
        float v = fmaxf(fmaxf(rowp[2 * wp], rowp[2 * wp + 1]),
                        fmaxf(rowp[56 + 2 * wp], rowp[56 + 2 * wp + 1]));
        c3pB[(size_t)(mtile * 16 + m) * KP + chunk * 28 + wp] = (bf16_t)v;
    }
    if (chunk == 0) {       // zero the K padding 784..799 once per (b, mtile)
        for (int i = tid; i < 16 * 16; i += 128) {
            const int m = i / 16, cc = 784 + (i % 16);
            c3pB[(size_t)(mtile * 16 + m) * KP + cc] = (bf16_t)0.f;
        }
    }
}

// ---------------------------------------------------------------------------
// Kernel 3: fused attention. Per (batch, 16-row n-block):
//   scores[16 x 784] via f32 WMMA (K=16) kept in registers,
//   softmax (shfl + LDS reductions), attn -> LDS bf16 [16 x KP],
//   refined[256 x 16] = c3p[256 x KP] @ attn^T via bf16 WMMA,
//   out = gamma * refined + x.
// ---------------------------------------------------------------------------
__global__ __launch_bounds__(256) void attn_kernel(
    const float* __restrict__ x, const float* __restrict__ c2,
    const float* __restrict__ c1p, const bf16_t* __restrict__ c3p,
    const float* __restrict__ gamma, float* __restrict__ out) {
    __shared__ alignas(16) bf16_t attn[16 * KP];    // 25.6 KB
    __shared__ float partial[8 * 16];
    __shared__ float rowstat[16];

    const int b = blockIdx.y, n0 = blockIdx.x * 16;
    const int tid = threadIdx.x, lane = tid & 31, wave = tid >> 5;
    const int nl = lane & 15, hh = lane >> 4;
    const float* c2B  = c2  + (size_t)b * CR * HW;
    const float* c1pB = c1p + (size_t)b * CR * HWP;

    // ---- phase 1: scores (M = n rows, N = m cols, K = 16 channels) ----
    v2f afr[4];
#pragma unroll
    for (int ks = 0; ks < 4; ++ks)
#pragma unroll
        for (int e = 0; e < 2; ++e)
            afr[ks][e] = c2B[(size_t)(ks * 4 + 2 * hh + e) * HW + n0 + nl];

    v8f sc[7];                                      // up to 7 m-tiles per wave (49 total)
#pragma unroll
    for (int s = 0; s < 7; ++s) {
        const int mt = wave + 8 * s;
        if (mt >= 49) break;
        v8f a = zero8();
#pragma unroll
        for (int ks = 0; ks < 4; ++ks) {
            v2f bfr;
#pragma unroll
            for (int e = 0; e < 2; ++e)
                bfr[e] = c1pB[(size_t)(ks * 4 + 2 * hh + e) * HWP + mt * 16 + nl];
            a = __builtin_amdgcn_wmma_f32_16x16x4_f32(
                false, afr[ks], false, bfr, (short)0, a, false, false);
        }
        sc[s] = a;
    }

    // ---- softmax over 784 (rows distributed: lane holds 8 rows x 1 col per tile) ----
    float lmax[8];
#pragma unroll
    for (int r = 0; r < 8; ++r) lmax[r] = -3.0e38f;
#pragma unroll
    for (int s = 0; s < 7; ++s) {
        if (wave + 8 * s >= 49) break;
#pragma unroll
        for (int r = 0; r < 8; ++r) lmax[r] = fmaxf(lmax[r], sc[s][r]);
    }
#pragma unroll
    for (int m = 1; m < 16; m <<= 1)
#pragma unroll
        for (int r = 0; r < 8; ++r) lmax[r] = fmaxf(lmax[r], __shfl_xor(lmax[r], m));
    if (nl == 0)
#pragma unroll
        for (int r = 0; r < 8; ++r) partial[wave * 16 + r + 8 * hh] = lmax[r];
    __syncthreads();
    if (tid < 16) {
        float v = partial[tid];
#pragma unroll
        for (int w = 1; w < 8; ++w) v = fmaxf(v, partial[w * 16 + tid]);
        rowstat[tid] = v;
    }
    __syncthreads();
    float rmax[8], lsum[8];
#pragma unroll
    for (int r = 0; r < 8; ++r) { rmax[r] = rowstat[r + 8 * hh]; lsum[r] = 0.f; }
#pragma unroll
    for (int s = 0; s < 7; ++s) {
        if (wave + 8 * s >= 49) break;
#pragma unroll
        for (int r = 0; r < 8; ++r) {
            const float e = __expf(sc[s][r] - rmax[r]);
            sc[s][r] = e;
            lsum[r] += e;
        }
    }
#pragma unroll
    for (int m = 1; m < 16; m <<= 1)
#pragma unroll
        for (int r = 0; r < 8; ++r) lsum[r] += __shfl_xor(lsum[r], m);
    if (nl == 0)
#pragma unroll
        for (int r = 0; r < 8; ++r) partial[wave * 16 + r + 8 * hh] = lsum[r];
    __syncthreads();
    if (tid < 16) {
        float v = 0.f;
#pragma unroll
        for (int w = 0; w < 8; ++w) v += partial[w * 16 + tid];
        rowstat[tid] = 1.f / v;
    }
    {   // zero K padding of attn (cols 784..799)
        const int rr = tid >> 4, cc = tid & 15;
        attn[(size_t)rr * KP + 784 + cc] = (bf16_t)0.f;
    }
    __syncthreads();
    float rinv[8];
#pragma unroll
    for (int r = 0; r < 8; ++r) rinv[r] = rowstat[r + 8 * hh];
#pragma unroll
    for (int s = 0; s < 7; ++s) {
        const int mt = wave + 8 * s;
        if (mt >= 49) break;
#pragma unroll
        for (int r = 0; r < 8; ++r)
            attn[(size_t)(r + 8 * hh) * KP + mt * 16 + nl] = (bf16_t)(sc[s][r] * rinv[r]);
    }
    __syncthreads();

    // ---- phase 2: refined[256 x 16] = c3p @ attn^T, bf16 WMMA, K = KP ----
    const bf16_t* c3pB = c3p + (size_t)b * C * KP;
    const float g = gamma[0];
#pragma unroll
    for (int ct = 0; ct < 2; ++ct) {
        const int ctile = wave * 2 + ct;            // 16 channel tiles over 8 waves
        v8f a = zero8();
        const bf16_t* arow = c3pB + (size_t)(ctile * 16 + nl) * KP;
        for (int k0 = 0; k0 < KP; k0 += 32) {
            v16bf af  = load_a_row(arow, k0, lane);
            v16bf bfr = load_b_attn(attn, k0, lane);
            a = __builtin_amdgcn_wmma_f32_16x16x32_bf16(
                false, af, false, bfr, (short)0, a, false, false);
        }
#pragma unroll
        for (int r = 0; r < 8; ++r) {
            const int c = ctile * 16 + r + 8 * hh;
            const size_t oi = (size_t)b * C * HW + (size_t)c * HW + n0 + nl;
            out[oi] = g * a[r] + x[oi];
        }
    }
}

// ---------------------------------------------------------------------------
extern "C" void kernel_launch(void* const* d_in, const int* in_sizes, int n_in,
                              void* d_out, int out_size, void* d_ws, size_t ws_size,
                              hipStream_t stream) {
    (void)in_sizes; (void)n_in; (void)out_size; (void)ws_size;
    const float* x    = (const float*)d_in[0];
    const float* w1   = (const float*)d_in[1];
    const float* bn1w = (const float*)d_in[2];
    const float* bn1b = (const float*)d_in[3];
    const float* bn1m = (const float*)d_in[4];
    const float* bn1v = (const float*)d_in[5];
    const float* p1   = (const float*)d_in[6];
    const float* w2   = (const float*)d_in[7];
    const float* bn2w = (const float*)d_in[8];
    const float* bn2b = (const float*)d_in[9];
    const float* bn2m = (const float*)d_in[10];
    const float* bn2v = (const float*)d_in[11];
    const float* p2   = (const float*)d_in[12];
    const float* w3   = (const float*)d_in[13];
    const float* bn3w = (const float*)d_in[14];
    const float* bn3b = (const float*)d_in[15];
    const float* bn3m = (const float*)d_in[16];
    const float* bn3v = (const float*)d_in[17];
    const float* p3   = (const float*)d_in[18];
    const float* gam  = (const float*)d_in[19];

    char* ws = (char*)d_ws;
    float*  c2  = (float*)(ws + OFF_C2);
    float*  c1p = (float*)(ws + OFF_C1P);
    bf16_t* c3p = (bf16_t*)(ws + OFF_C3P);
    float*  out = (float*)d_out;

    dim3 g1(HW / 112, BATCH);                 // 28 x 16
    conv12_kernel<<<g1, 128, 0, stream>>>(x, w1, w2, bn1w, bn1b, bn1m, bn1v, p1,
                                          bn2w, bn2b, bn2m, bn2v, p2, c2, c1p);

    dim3 g2(HW / 112, C / 16, BATCH);         // 28 x 16 x 16
    conv3_kernel<<<g2, 128, 0, stream>>>(x, w3, bn3w, bn3b, bn3m, bn3v, p3, c3p);

    dim3 g3(HW / 16, BATCH);                  // 196 x 16
    attn_kernel<<<g3, 256, 0, stream>>>(x, c2, c1p, c3p, gam, out);
}
